// SlotAttention_10041633538130
// MI455X (gfx1250) — matrix operand order repaired and checked by reading, not compiled, not run on hardware
//
#include <hip/hip_runtime.h>
#include <hip/hip_bf16.h>
#include <math.h>

#define B_ 64
#define N_ 4096
#define D_ 256
#define S_ 8
#define NH_ 8
#define DK_ 32
#define ITERS_ 3
#define MLPH_ 512
#define EPS_ 1e-5f

typedef __attribute__((ext_vector_type(16))) __bf16 v16bf;
typedef __attribute__((ext_vector_type(8)))  float  v8f;
typedef __attribute__((ext_vector_type(4)))  unsigned int v4u;
typedef __attribute__((ext_vector_type(8)))  int v8i;
typedef __attribute__((ext_vector_type(4)))  int v4i;

#if defined(__HIP_DEVICE_COMPILE__) && __has_builtin(__builtin_amdgcn_tensor_load_to_lds) && __has_builtin(__builtin_amdgcn_s_wait_tensorcnt)
#define HAVE_TDM 1
#else
#define HAVE_TDM 0
#endif

union BF16Frag { v16bf v; unsigned short u[16]; };

__device__ __forceinline__ unsigned short f2bf(float x) {
  unsigned int u = __float_as_uint(x);
  u += 0x7FFFu + ((u >> 16) & 1u);
  return (unsigned short)(u >> 16);
}

// A-matrix 16x32 bf16 (ISA 7.12.2): lanes 0-15: K=0..7,16..23 ; lanes 16-31: K=8..15,24..31
__device__ __forceinline__ int frag_ka(int j, int half) {
  return (j < 8) ? (half * 8 + j) : (16 + half * 8 + (j - 8));
}
// B-matrix 32x16 bf16: lanes 0-15: K=0..7,16..23 ; lanes 16-31: K=8..15,24..31
__device__ __forceinline__ int frag_kb(int j, int half) {
  return (j < 8) ? (half * 8 + j) : (16 + half * 8 + (j - 8));
}

__device__ __forceinline__ v8f wmma_bf16(const v16bf& a, const v16bf& b, v8f c) {
  return __builtin_amdgcn_wmma_f32_16x16x32_bf16(false, a, false, b, (short)0, c, false, false);
}

// ---------------- weight conversion: wk/wv -> bf16 transposed [col][k] ----------------
__global__ __launch_bounds__(256) void wconv_kernel(const float* __restrict__ wk,
                                                    const float* __restrict__ wv,
                                                    unsigned short* __restrict__ wkT,
                                                    unsigned short* __restrict__ wvT) {
  int idx = blockIdx.x * 256 + threadIdx.x;   // 0..65535
  int col = idx >> 8;
  int k   = idx & 255;
  wkT[col * D_ + k] = f2bf(wk[k * D_ + col]);
  wvT[col * D_ + k] = f2bf(wv[k * D_ + col]);
}

// ---------------- slots init: mu + exp(log_sigma)*noise ----------------
__global__ __launch_bounds__(256) void slots_init_kernel(const float* __restrict__ noise,
                                                         const float* __restrict__ mu,
                                                         const float* __restrict__ ls,
                                                         float* __restrict__ slots) {
  int idx = blockIdx.x * 256 + threadIdx.x;
  int d = idx & (D_ - 1);
  slots[idx] = mu[d] + __expf(ls[d]) * noise[idx];
}

// ---------------- fused LayerNorm + K/V projection (TDM tile load + WMMA bf16) ----------------
// Each block: 64 rows of x. 8 waves: wave = rtile(0..3) x {K,V}
__global__ __launch_bounds__(256) void ln_kv_kernel(
    const float* __restrict__ x, const float* __restrict__ g_in, const float* __restrict__ b_in,
    const unsigned short* __restrict__ wkT, const unsigned short* __restrict__ wvT,
    const float* __restrict__ bk, const float* __restrict__ bv,
    unsigned short* __restrict__ Kbf, unsigned short* __restrict__ Vbf) {
  // fp32 tile and bf16 normalized tile alias: every fp32 read is register-buffered
  // before the first barrier, bf16 writes happen two barriers later.
  __shared__ union {
    float          xraw[64][D_];        // 64KB (TDM destination, contiguous)
    unsigned short xln[64][D_ + 8];     // 33KB, 16B-aligned row stride (528B)
  } sm;
  __shared__ float psum[256];
  __shared__ float psq[256];
  __shared__ float mrs[64][2];

  const int t = threadIdx.x;
  const int row = t >> 2;       // 0..63
  const int qtr = t & 3;        // 0..3
  const long long gbase = (long long)blockIdx.x * 64;

  float vals[64];

#if HAVE_TDM
  // --- Tensor Data Mover: DMA the 64x256 f32 tile into LDS (one op per block) ---
  if (t < 32) {
    const unsigned lds_off = (unsigned)(unsigned long long)(&sm.xraw[0][0]);
    const unsigned long long ga = (unsigned long long)(const void*)(x + gbase * D_);
    // D# group0: count=1 | lds_addr | global_addr[31:0] | global_addr[56:32], type=2
    v4u g0 = { 1u,
               lds_off,
               (unsigned)(ga & 0xFFFFFFFFu),
               (unsigned)(((ga >> 32) & 0x1FFFFFFu) | (2u << 30)) };
    // D# group1: data_size=4B(2); tensor_dim0=256; tensor_dim1=B*N; tile 256x64; dim0_stride=256
    v8i g1 = { (int)(2u << 16),                     // workgroup_mask=0, data_size=2(4B)
               (int)(256u << 16),                   // tensor_dim0[15:0] in [63:48]
               0,                                   // tensor_dim0[31:16]=0, tensor_dim1[15:0]=0
               (int)(4u | (256u << 16)),            // tensor_dim1[31:16]=4 (262144), tile_dim0=256
               64,                                  // tile_dim1=64, tile_dim2=0
               256,                                 // tensor_dim0_stride[31:0]
               0, 0 };
    v4i z4 = {0, 0, 0, 0};
#if defined(__clang_major__) && (__clang_major__ >= 23)
    v8i z8 = {0, 0, 0, 0, 0, 0, 0, 0};
    __builtin_amdgcn_tensor_load_to_lds(g0, g1, z4, z4, z8, 0);
#else
    __builtin_amdgcn_tensor_load_to_lds(g0, g1, z4, z4, 0);
#endif
  }
  __builtin_amdgcn_s_wait_tensorcnt(0);
  __syncthreads();
#pragma unroll
  for (int i = 0; i < 64; ++i) vals[i] = sm.xraw[row][qtr * 64 + i];
#else
  {
    const float* xrow = x + (gbase + row) * D_ + qtr * 64;
#pragma unroll
    for (int i = 0; i < 64; ++i) vals[i] = xrow[i];
  }
#endif

  float s = 0.f, sq = 0.f;
#pragma unroll
  for (int i = 0; i < 64; ++i) { float v = vals[i]; s += v; sq += v * v; }
  psum[t] = s; psq[t] = sq;
  __syncthreads();
  if (qtr == 0) {
    float ss = psum[t] + psum[t + 1] + psum[t + 2] + psum[t + 3];
    float qq = psq[t]  + psq[t + 1]  + psq[t + 2]  + psq[t + 3];
    float mean = ss * (1.0f / D_);
    float var  = qq * (1.0f / D_) - mean * mean;
    mrs[row][0] = mean;
    mrs[row][1] = rsqrtf(var + EPS_);
  }
  __syncthreads();
  {
    float mean = mrs[row][0], rstd = mrs[row][1];
#pragma unroll
    for (int i = 0; i < 64; ++i) {
      int c = qtr * 64 + i;
      sm.xln[row][c] = f2bf((vals[i] - mean) * rstd * g_in[c] + b_in[c]);
    }
  }
  __syncthreads();

  const int wave = t >> 5;          // 0..7
  const int lane = t & 31;
  const int half = lane >> 4;
  const int lr   = lane & 15;
  const int rtile = wave & 3;       // 16-row tile
  const int sel   = wave >> 2;      // 0 = K, 1 = V
  const unsigned short* wT = sel ? wvT : wkT;
  const float* bias = sel ? bv : bk;
  unsigned short* outp = sel ? Vbf : Kbf;

  // preload A fragments for all 8 K-chunks (rows fixed per wave)
  BF16Frag afr[8];
#pragma unroll
  for (int kk = 0; kk < 8; ++kk)
#pragma unroll
    for (int j = 0; j < 16; ++j)
      afr[kk].u[j] = sm.xln[rtile * 16 + lr][kk * 32 + frag_ka(j, half)];

  for (int nt = 0; nt < 16; ++nt) {
    const int col = nt * 16 + lr;
    const unsigned short* wcol = wT + col * D_;
    // preload the whole weight column (all 8 B fragments) so loads overlap WMMAs
    BF16Frag bfr[8];
#pragma unroll
    for (int kk = 0; kk < 8; ++kk)
#pragma unroll
      for (int j = 0; j < 16; ++j) bfr[kk].u[j] = wcol[kk * 32 + frag_kb(j, half)];
    v8f acc = {};
#pragma unroll
    for (int kk = 0; kk < 8; ++kk) acc = wmma_bf16(afr[kk].v, bfr[kk].v, acc);

    const float bb = bias[col];
    const int hh = col >> 5, dk = col & 31;
#pragma unroll
    for (int v = 0; v < 8; ++v) {
      int rl = rtile * 16 + v + half * 8;
      long long g = gbase + rl;
      int bidx = (int)(g >> 12);          // g / N_
      int n    = (int)(g & (N_ - 1));
      outp[(((long long)(bidx * NH_ + hh)) * N_ + n) * DK_ + dk] = f2bf(acc[v] + bb);
    }
  }
}

// ---------------- slot LN + Q projection (fp32, tiny) ----------------
__global__ __launch_bounds__(256) void qproj_kernel(
    const float* __restrict__ slots, const float* __restrict__ g_slot, const float* __restrict__ b_slot,
    const float* __restrict__ wq, const float* __restrict__ bq,
    unsigned short* __restrict__ Qbf) {
  __shared__ float sp[D_];
  __shared__ float red[256];
  const int t = threadIdx.x;
  const int row = blockIdx.x;       // b*S + s
  const int b = row / S_, s = row % S_;
  const float v = slots[(long long)row * D_ + t];

  red[t] = v; __syncthreads();
  for (int st = 128; st > 0; st >>= 1) { if (t < st) red[t] += red[t + st]; __syncthreads(); }
  const float mean = red[0] * (1.0f / D_);
  __syncthreads();
  red[t] = (v - mean) * (v - mean); __syncthreads();
  for (int st = 128; st > 0; st >>= 1) { if (t < st) red[t] += red[t + st]; __syncthreads(); }
  const float rstd = rsqrtf(red[0] * (1.0f / D_) + EPS_);
  __syncthreads();
  sp[t] = (v - mean) * rstd * g_slot[t] + b_slot[t];
  __syncthreads();

  float acc = bq[t];
  for (int k = 0; k < D_; ++k) acc = fmaf(sp[k], wq[k * D_ + t], acc);
  const int h = t >> 5, dk = t & 31;
  Qbf[(((long long)(b * NH_ + h)) * 16 + s) * DK_ + dk] = f2bf(acc);
}

// ---------------- flash attention per (b,h): one wave, online softmax, double-buffered ----------------
struct KV4 { BF16Frag k0, k1, v0, v1; };

__device__ __forceinline__ void load_kv(const unsigned short* __restrict__ kb,
                                        const unsigned short* __restrict__ vb,
                                        int n0, int lr, int half, KV4& f) {
#pragma unroll
  for (int j = 0; j < 16; ++j) {
    int kj = frag_kb(j, half);
    f.k0.u[j] = kb[(long long)(n0 + lr) * DK_ + kj];        // B = K^T (dk major per lane-half)
    f.k1.u[j] = kb[(long long)(n0 + 16 + lr) * DK_ + kj];
    f.v0.u[j] = vb[(long long)(n0 + kj) * DK_ + lr];        // B = V (keys = K-dim)
    f.v1.u[j] = vb[(long long)(n0 + kj) * DK_ + 16 + lr];
  }
}

__global__ __launch_bounds__(32) void attn_kernel(
    const unsigned short* __restrict__ Qbf, const unsigned short* __restrict__ Kbf,
    const unsigned short* __restrict__ Vbf, float* __restrict__ attn_out) {
  __shared__ unsigned short pshared[16][36];
  const int lane = threadIdx.x;
  const int half = lane >> 4, lr = lane & 15;
  const int bh = blockIdx.x;
  const int b = bh / NH_, h = bh % NH_;
  const unsigned short* qb = Qbf + (long long)bh * 16 * DK_;
  const unsigned short* kb = Kbf + (long long)bh * N_ * DK_;
  const unsigned short* vb = Vbf + (long long)bh * N_ * DK_;

  BF16Frag qf;
#pragma unroll
  for (int j = 0; j < 16; ++j) qf.u[j] = qb[lr * DK_ + frag_ka(j, half)];

  const float scale = 0.1767766953f;  // 1/sqrt(DK)
  float m[8], l[8];
  v8f o0 = {}, o1 = {};
#pragma unroll
  for (int v = 0; v < 8; ++v) { m[v] = -1e30f; l[v] = 0.f; }

  auto compute_chunk = [&](const KV4& f) {
    v8f c0 = {}, c1 = {};
    c0 = wmma_bf16(qf.v, f.k0.v, c0);
    c1 = wmma_bf16(qf.v, f.k1.v, c1);

    float p0[8], p1[8];
#pragma unroll
    for (int v = 0; v < 8; ++v) {
      float s0 = c0[v] * scale, s1 = c1[v] * scale;
      float mx = fmaxf(s0, s1);
#pragma unroll
      for (int msk = 8; msk >= 1; msk >>= 1) mx = fmaxf(mx, __shfl_xor(mx, msk, 32));
      float mnew = fmaxf(m[v], mx);
      p0[v] = __expf(s0 - mnew);
      p1[v] = __expf(s1 - mnew);
      float ps = p0[v] + p1[v];
#pragma unroll
      for (int msk = 8; msk >= 1; msk >>= 1) ps += __shfl_xor(ps, msk, 32);
      float corr = __expf(m[v] - mnew);
      l[v] = l[v] * corr + ps;
      o0[v] *= corr;
      o1[v] *= corr;
      m[v] = mnew;
    }
    // C-layout -> A-layout through LDS
#pragma unroll
    for (int v = 0; v < 8; ++v) {
      int r = v + half * 8;
      pshared[r][lr]      = f2bf(p0[v]);
      pshared[r][16 + lr] = f2bf(p1[v]);
    }
    __syncthreads();
    BF16Frag pf;
#pragma unroll
    for (int j = 0; j < 16; ++j) pf.u[j] = pshared[lr][frag_ka(j, half)];
    __syncthreads();
    o0 = wmma_bf16(pf.v, f.v0.v, o0);
    o1 = wmma_bf16(pf.v, f.v1.v, o1);
  };

  // ping-pong double buffering over 64-key super-chunks (N_/64 = 64 iterations)
  KV4 bufA, bufB;
  load_kv(kb, vb, 0, lr, half, bufA);
  for (int n0 = 0; n0 < N_; n0 += 64) {
    load_kv(kb, vb, n0 + 32, lr, half, bufB);
    if (n0 + 64 < N_) {
      __builtin_prefetch(kb + (long long)(n0 + 64) * DK_, 0, 0);
      __builtin_prefetch(vb + (long long)(n0 + 64) * DK_, 0, 0);
    }
    compute_chunk(bufA);
    if (n0 + 64 < N_) load_kv(kb, vb, n0 + 64, lr, half, bufA);
    compute_chunk(bufB);
  }

  if (half == 0) {
#pragma unroll
    for (int v = 0; v < S_; ++v) {        // valid slot rows 0..7 live in lanes 0-15
      float inv = 1.0f / l[v];
      long long base = ((long long)(b * S_ + v)) * D_ + h * DK_;
      attn_out[base + lr]      = o0[v] * inv;
      attn_out[base + 16 + lr] = o1[v] * inv;
    }
  }
}

// ---------------- wo proj + GRU + LN + residual MLP (fp32, 512 rows) ----------------
__global__ __launch_bounds__(256) void update_kernel(
    const float* __restrict__ attn_out, const float* __restrict__ slots_in,
    const float* __restrict__ wo, const float* __restrict__ bo,
    const float* __restrict__ w_ih, const float* __restrict__ b_ih,
    const float* __restrict__ w_hh, const float* __restrict__ b_hh,
    const float* __restrict__ mw1, const float* __restrict__ mb1,
    const float* __restrict__ mw2, const float* __restrict__ mb2,
    const float* __restrict__ g_mlp, const float* __restrict__ b_mlp,
    float* __restrict__ slots_out) {
  __shared__ float oa[D_];
  __shared__ float hr[D_];
  __shared__ float o2[D_];
  __shared__ float mn[D_];
  __shared__ float hid[MLPH_];
  __shared__ float red[256];
  const int t = threadIdx.x;
  const long long row = blockIdx.x;

  oa[t] = attn_out[row * D_ + t];
  hr[t] = slots_in[row * D_ + t];
  __syncthreads();

  // out = o @ wo + bo
  float acc = bo[t];
  for (int k = 0; k < D_; ++k) acc = fmaf(oa[k], wo[k * D_ + t], acc);
  o2[t] = acc;
  __syncthreads();

  // GRU gates
  float gir = b_ih[t], giz = b_ih[D_ + t], gin = b_ih[2 * D_ + t];
  {
    const float* wr = w_ih + (long long)t * D_;
    const float* wz = w_ih + (long long)(D_ + t) * D_;
    const float* wn = w_ih + (long long)(2 * D_ + t) * D_;
    for (int k = 0; k < D_; ++k) {
      float xv = o2[k];
      gir = fmaf(xv, wr[k], gir);
      giz = fmaf(xv, wz[k], giz);
      gin = fmaf(xv, wn[k], gin);
    }
  }
  float ghr = b_hh[t], ghz = b_hh[D_ + t], ghn = b_hh[2 * D_ + t];
  {
    const float* vr = w_hh + (long long)t * D_;
    const float* vz = w_hh + (long long)(D_ + t) * D_;
    const float* vn = w_hh + (long long)(2 * D_ + t) * D_;
    for (int k = 0; k < D_; ++k) {
      float hv = hr[k];
      ghr = fmaf(hv, vr[k], ghr);
      ghz = fmaf(hv, vz[k], ghz);
      ghn = fmaf(hv, vn[k], ghn);
    }
  }
  const float r  = 1.0f / (1.0f + __expf(-(gir + ghr)));
  const float z  = 1.0f / (1.0f + __expf(-(giz + ghz)));
  const float ng = tanhf(gin + r * ghn);
  const float snew = (1.0f - z) * ng + z * hr[t];
  __syncthreads();

  // LayerNorm(snew)
  red[t] = snew; __syncthreads();
  for (int st = 128; st > 0; st >>= 1) { if (t < st) red[t] += red[t + st]; __syncthreads(); }
  const float mean = red[0] * (1.0f / D_);
  __syncthreads();
  red[t] = (snew - mean) * (snew - mean); __syncthreads();
  for (int st = 128; st > 0; st >>= 1) { if (t < st) red[t] += red[t + st]; __syncthreads(); }
  const float rstd = rsqrtf(red[0] * (1.0f / D_) + EPS_);
  __syncthreads();
  mn[t] = (snew - mean) * rstd * g_mlp[t] + b_mlp[t];
  __syncthreads();

  // MLP layer 1 (two cols per thread)
  float a1 = mb1[t], a2 = mb1[t + 256];
  for (int k = 0; k < D_; ++k) {
    float mv = mn[k];
    a1 = fmaf(mv, mw1[k * MLPH_ + t], a1);
    a2 = fmaf(mv, mw1[k * MLPH_ + t + 256], a2);
  }
  hid[t] = fmaxf(a1, 0.f);
  hid[t + 256] = fmaxf(a2, 0.f);
  __syncthreads();

  // MLP layer 2 + residual
  float out = snew + mb2[t];
  for (int k = 0; k < MLPH_; ++k) out = fmaf(hid[k], mw2[k * D_ + t], out);
  slots_out[row * D_ + t] = out;
}

extern "C" void kernel_launch(void* const* d_in, const int* in_sizes, int n_in,
                              void* d_out, int out_size, void* d_ws, size_t ws_size,
                              hipStream_t stream) {
  (void)in_sizes; (void)n_in; (void)out_size; (void)ws_size;
  const float* inputs = (const float*)d_in[0];
  const float* noise  = (const float*)d_in[1];
  const float* mu     = (const float*)d_in[2];
  const float* ls     = (const float*)d_in[3];
  const float* wq = (const float*)d_in[4];  const float* bq = (const float*)d_in[5];
  const float* wk = (const float*)d_in[6];  const float* bk = (const float*)d_in[7];
  const float* wv = (const float*)d_in[8];  const float* bv = (const float*)d_in[9];
  const float* wo = (const float*)d_in[10]; const float* bo = (const float*)d_in[11];
  const float* w_ih = (const float*)d_in[12]; const float* b_ih = (const float*)d_in[13];
  const float* w_hh = (const float*)d_in[14]; const float* b_hh = (const float*)d_in[15];
  const float* mw1 = (const float*)d_in[16]; const float* mb1 = (const float*)d_in[17];
  const float* mw2 = (const float*)d_in[18]; const float* mb2 = (const float*)d_in[19];
  const float* g_in = (const float*)d_in[20];   const float* b_in = (const float*)d_in[21];
  const float* g_slot = (const float*)d_in[22]; const float* b_slot = (const float*)d_in[23];
  const float* g_mlp = (const float*)d_in[24];  const float* b_mlp = (const float*)d_in[25];

  char* ws = (char*)d_ws;
  size_t off = 0;
  unsigned short* Kbf = (unsigned short*)(ws + off); off += (size_t)B_ * NH_ * N_ * DK_ * 2;
  unsigned short* Vbf = (unsigned short*)(ws + off); off += (size_t)B_ * NH_ * N_ * DK_ * 2;
  unsigned short* Qbf = (unsigned short*)(ws + off); off += (size_t)B_ * NH_ * 16 * DK_ * 2;
  float* slots    = (float*)(ws + off); off += (size_t)B_ * S_ * D_ * 4;
  float* attn_out = (float*)(ws + off); off += (size_t)B_ * S_ * D_ * 4;
  unsigned short* wkT = (unsigned short*)(ws + off); off += (size_t)D_ * D_ * 2;
  unsigned short* wvT = (unsigned short*)(ws + off); off += (size_t)D_ * D_ * 2;

  wconv_kernel<<<256, 256, 0, stream>>>(wk, wv, wkT, wvT);
  slots_init_kernel<<<(B_ * S_ * D_) / 256, 256, 0, stream>>>(noise, mu, ls, slots);
  ln_kv_kernel<<<(B_ * N_) / 64, 256, 0, stream>>>(inputs, g_in, b_in, wkT, wvT, bk, bv, Kbf, Vbf);

  for (int it = 0; it < ITERS_; ++it) {
    qproj_kernel<<<B_ * S_, 256, 0, stream>>>(slots, g_slot, b_slot, wq, bq, Qbf);
    attn_kernel<<<B_ * NH_, 32, 0, stream>>>(Qbf, Kbf, Vbf, attn_out);
    float* dst = (it == ITERS_ - 1) ? (float*)d_out : slots;
    update_kernel<<<B_ * S_, 256, 0, stream>>>(attn_out, slots, wo, bo,
                                               w_ih, b_ih, w_hh, b_hh,
                                               mw1, mb1, mw2, mb2,
                                               g_mlp, b_mlp, dst);
  }
}
